// MambaBlock_89249420411545
// MI455X (gfx1250) — matrix operand order repaired and checked by reading, not compiled, not run on hardware
//
#include <hip/hip_runtime.h>
#include <hip/hip_bf16.h>
#include <math.h>
#include <stdint.h>

// ---------------------------------------------------------------------------
// Mamba block forward for gfx1250 (MI455X).
//   GEMMs : v_wmma_f32_16x16x32_bf16, 32x64 tile/wave, ping-pong K pipeline
//           (no conditional loads, no buffer-rotation movs)
//   Scan  : register-resident 16-state recurrence; B/C staged into LDS with
//           global_load_async_to_lds_b128 + s_wait_asynccnt (CDNA5 async path)
// ---------------------------------------------------------------------------

typedef __bf16 v16bf __attribute__((ext_vector_type(16)));
typedef __bf16 v8bf  __attribute__((ext_vector_type(8)));
typedef float  v8f   __attribute__((ext_vector_type(8)));

#define D_MODEL 1024
#define D_STATE 16
#define D_INNER 2048
#define DT_RANK 64
#define KSIZE   4
#define BATCH   2
#define SEQLEN  2048
#define MROWS   (BATCH * SEQLEN)          // 4096 token rows

// ---- workspace layout (bytes) ---------------------------------------------
#define XB_OFF     ((size_t)0)                                  // x bf16 [4096 x 1024]
#define WINT_OFF   (XB_OFF    + (size_t)MROWS*D_MODEL*2)        // w_in^T bf16 [4096 x 1024]
#define WOUTT_OFF  (WINT_OFF  + (size_t)(2*D_INNER)*D_MODEL*2)  // w_out^T bf16 [1024 x 2048]
#define XZ_OFF     (WOUTT_OFF + (size_t)D_MODEL*D_INNER*2)      // xz f32 [4096 x 4096]
#define UACT_OFF   (XZ_OFF    + (size_t)MROWS*2*D_INNER*4)      // u_act f32 [4096 x 2048]
#define XDBL_OFF   (UACT_OFF  + (size_t)MROWS*D_INNER*4)        // x_dbl f32 [4096 x 96]
#define DTB_OFF    (XDBL_OFF  + (size_t)MROWS*96*4)             // dt f32 [4096 x 2048]
#define YG_OFF     (DTB_OFF   + (size_t)MROWS*D_INNER*4)        // gated y bf16 [4096 x 2048]

// ---- CDNA5 async-counter wait ---------------------------------------------
#if defined(__has_builtin)
#if __has_builtin(__builtin_amdgcn_s_wait_asynccnt)
#define WAIT_ASYNC() __builtin_amdgcn_s_wait_asynccnt(0)
#endif
#endif
#ifndef WAIT_ASYNC
#define WAIT_ASYNC() asm volatile("s_wait_asynccnt 0x0" ::: "memory")
#endif

// ---------------------------------------------------------------------------
// fp32 -> bf16 copy
__global__ __launch_bounds__(256)
void k_cvt_bf16(const float* __restrict__ src, __bf16* __restrict__ dst, int n) {
  int i = blockIdx.x * 256 + threadIdx.x;
  if (i < n) dst[i] = (__bf16)src[i];
}

// fp32 (K x N) -> bf16 transposed (N x K)
__global__ __launch_bounds__(256)
void k_cvtT_bf16(const float* __restrict__ src, __bf16* __restrict__ dst, int K, int N) {
  int i = blockIdx.x * 256 + threadIdx.x;
  if (i < K * N) {
    int nrow = i / K;
    int k    = i % K;
    dst[i] = (__bf16)src[(size_t)k * N + nrow];
  }
}

// ---------------------------------------------------------------------------
// bf16 WMMA GEMM: C[M,N] = A[M,K] * Bt[N,K]^T   (Bt row-major N x K)
// One wave computes a 32x64 tile (2 M-frags x 4 N-frags = 8 WMMAs / 32-deep
// K-step).  Ping-pong fragment buffers, written in place; K-loop has no
// conditionals, so loads for step k+1 stay in flight across the WMMAs of
// step k.  256-thread blocks = 8 waves; grid exact -> EXEC all-1s.
__global__ __launch_bounds__(256)
void k_gemm_bf16_wmma(const __bf16* __restrict__ A, const __bf16* __restrict__ Bt,
                      float* __restrict__ C, int M, int N, int K) {
  const int lane  = threadIdx.x & 31;
  const int wave  = blockIdx.x * 8 + (threadIdx.x >> 5);
  const int tilesN = N >> 6;
  const int tm = wave / tilesN;               // 32-row strip
  const int tn = wave % tilesN;               // 64-col strip
  const int lm = lane & 15;
  const int hl = lane >> 4;                   // lane half
  const int kbA = hl << 4;                    // A K-base: 0 / 16
  const int kbB = hl << 4;                    // B K-base: 0 / 16

  const __bf16* a0 = A  + (size_t)(tm * 32 + lm) * K + kbA;
  const __bf16* a1 = a0 + (size_t)16 * K;
  const __bf16* b0 = Bt + (size_t)(tn * 64 + lm) * K + kbB;
  const size_t  bs = (size_t)16 * K;

  auto ldA = [](const __bf16* p) {
    v8bf lo = *reinterpret_cast<const v8bf*>(p);
    v8bf hi = *reinterpret_cast<const v8bf*>(p + 8);
    v16bf r;
#pragma unroll
    for (int i = 0; i < 8; ++i) { r[i] = lo[i]; r[i + 8] = hi[i]; }
    return r;
  };
  auto ldB = [](const __bf16* p) { return *reinterpret_cast<const v16bf*>(p); };

  v8f acc00 = {}, acc01 = {}, acc02 = {}, acc03 = {};
  v8f acc10 = {}, acc11 = {}, acc12 = {}, acc13 = {};

#define LOAD_FRAGS(A0f, A1f, B0f, B1f, B2f, B3f, kofs)                         \
  do {                                                                         \
    A0f = ldA(a0 + (kofs));  A1f = ldA(a1 + (kofs));                           \
    B0f = ldB(b0 + (kofs));  B1f = ldB(b0 + bs + (kofs));                      \
    B2f = ldB(b0 + 2 * bs + (kofs)); B3f = ldB(b0 + 3 * bs + (kofs));          \
  } while (0)

#define WMMA8(A0f, A1f, B0f, B1f, B2f, B3f)                                                              \
  do {                                                                                                   \
    acc00 = __builtin_amdgcn_wmma_f32_16x16x32_bf16(false, A0f, false, B0f, (short)0, acc00, false, false); \
    acc01 = __builtin_amdgcn_wmma_f32_16x16x32_bf16(false, A0f, false, B1f, (short)0, acc01, false, false); \
    acc02 = __builtin_amdgcn_wmma_f32_16x16x32_bf16(false, A0f, false, B2f, (short)0, acc02, false, false); \
    acc03 = __builtin_amdgcn_wmma_f32_16x16x32_bf16(false, A0f, false, B3f, (short)0, acc03, false, false); \
    acc10 = __builtin_amdgcn_wmma_f32_16x16x32_bf16(false, A1f, false, B0f, (short)0, acc10, false, false); \
    acc11 = __builtin_amdgcn_wmma_f32_16x16x32_bf16(false, A1f, false, B1f, (short)0, acc11, false, false); \
    acc12 = __builtin_amdgcn_wmma_f32_16x16x32_bf16(false, A1f, false, B2f, (short)0, acc12, false, false); \
    acc13 = __builtin_amdgcn_wmma_f32_16x16x32_bf16(false, A1f, false, B3f, (short)0, acc13, false, false); \
  } while (0)

  v16bf pa0, pa1, pb0, pb1, pb2, pb3;   // ping buffer
  v16bf qa0, qa1, qb0, qb1, qb2, qb3;   // pong buffer

  const int nIter = K >> 5;             // 32 or 64: even, >= 2
  int k = 32;

  LOAD_FRAGS(pa0, pa1, pb0, pb1, pb2, pb3, 0);        // step 0 -> ping

  for (int kk = 0; kk < nIter - 2; kk += 2) {
    LOAD_FRAGS(qa0, qa1, qb0, qb1, qb2, qb3, k);      // step kk+1 -> pong
    __builtin_prefetch(a0 + k + 64, 0, 1);            // global_prefetch_b8
    __builtin_prefetch(b0 + k + 64, 0, 1);
    k += 32;
    WMMA8(pa0, pa1, pb0, pb1, pb2, pb3);              // step kk
    LOAD_FRAGS(pa0, pa1, pb0, pb1, pb2, pb3, k);      // step kk+2 -> ping
    k += 32;
    WMMA8(qa0, qa1, qb0, qb1, qb2, qb3);              // step kk+1
  }

  // epilogue: ping holds step nIter-2; load final step into pong
  LOAD_FRAGS(qa0, qa1, qb0, qb1, qb2, qb3, k);
  WMMA8(pa0, pa1, pb0, pb1, pb2, pb3);
  WMMA8(qa0, qa1, qb0, qb1, qb2, qb3);

#undef LOAD_FRAGS
#undef WMMA8

  // store: VGPR r -> row (r + 8*lane_half), col = lane&15 per C/D layout
#pragma unroll
  for (int r = 0; r < 8; ++r) {
    int row0 = tm * 32 + hl * 8 + r;
    float* c0 = C + (size_t)row0 * N + tn * 64 + lm;
    float* c1 = c0 + (size_t)16 * N;
    c0[0]  = acc00[r]; c0[16] = acc01[r]; c0[32] = acc02[r]; c0[48] = acc03[r];
    c1[0]  = acc10[r]; c1[16] = acc11[r]; c1[32] = acc12[r]; c1[48] = acc13[r];
  }
}

// ---------------------------------------------------------------------------
// depthwise causal conv1d (K=4, left pad 3) + bias + SiLU on u = xz[:, :2048]
__global__ __launch_bounds__(256)
void k_conv_silu(const float* __restrict__ xz, const float* __restrict__ w_conv,
                 const float* __restrict__ b_conv, float* __restrict__ u_act) {
  int i   = blockIdx.x * 256 + threadIdx.x;   // over MROWS * D_INNER
  int d   = i & (D_INNER - 1);
  int row = i >> 11;                          // D_INNER = 2^11
  int b   = row >> 11;                        // SEQLEN  = 2^11
  int l   = row & (SEQLEN - 1);
  float acc = b_conv[d];
#pragma unroll
  for (int kk = 0; kk < KSIZE; ++kk) {
    int ll = l - (KSIZE - 1) + kk;
    if (ll >= 0)
      acc += w_conv[d * KSIZE + kk] * xz[(size_t)(b * SEQLEN + ll) * (2 * D_INNER) + d];
  }
  u_act[i] = acc * (1.0f / (1.0f + __expf(-acc)));       // silu
}

// ---------------------------------------------------------------------------
// x_dbl[row, c] = sum_k u_act[row, k] * w_x[k, c]    (N = 96: scalar dot)
__global__ __launch_bounds__(256)
void k_xdbl(const float* __restrict__ u_act, const float* __restrict__ w_x,
            float* __restrict__ x_dbl) {
  int i = blockIdx.x * 256 + threadIdx.x;     // over MROWS * 96
  if (i >= MROWS * 96) return;
  int row = i / 96, c = i % 96;
  const float* urow = u_act + (size_t)row * D_INNER;
  float acc = 0.f;
  for (int k = 0; k < D_INNER; ++k) acc += urow[k] * w_x[(size_t)k * 96 + c];
  x_dbl[i] = acc;
}

// ---------------------------------------------------------------------------
// dt[row, d] = softplus( x_dbl[row, :64] @ w_dt[:, d] + b_dt[d] )
__global__ __launch_bounds__(256)
void k_dt(const float* __restrict__ x_dbl, const float* __restrict__ w_dt,
          const float* __restrict__ b_dt, float* __restrict__ dt) {
  int i   = blockIdx.x * 256 + threadIdx.x;   // over MROWS * D_INNER
  int d   = i & (D_INNER - 1);
  int row = i >> 11;
  const float* xr = x_dbl + (size_t)row * 96;
  float acc = b_dt[d];
#pragma unroll
  for (int r = 0; r < DT_RANK; ++r) acc += xr[r] * w_dt[(size_t)r * D_INNER + d];
  dt[i] = (acc > 20.f) ? acc : log1pf(__expf(acc));      // softplus
}

// ---------------------------------------------------------------------------
// Selective scan over L, fused with +u*D, *silu(z), bf16 output for out-GEMM.
// One thread per (batch, channel); h[16] in registers.  The shared (B, C)
// block (32 floats / timestep, 16B-aligned at byte row*384+256) is staged in
// 64-timestep chunks via async global->LDS b128 copies.
__global__ __launch_bounds__(256)
void k_scan(const float* __restrict__ u_act, const float* __restrict__ dt,
            const float* __restrict__ x_dbl, const float* __restrict__ xz,
            const float* __restrict__ A_log, const float* __restrict__ D_par,
            __bf16* __restrict__ yg) {
  __shared__ float lds_bc[64 * 32];           // 8 KB: B(16)+C(16) x 64 steps

  const int b = blockIdx.x >> 3;              // 8 blocks per batch
  const int d = ((blockIdx.x & 7) << 8) + threadIdx.x;

  float a[D_STATE];
#pragma unroll
  for (int n = 0; n < D_STATE; ++n) a[n] = -__expf(A_log[d * D_STATE + n]);
  const float Dd = D_par[d];

  float h[D_STATE];
#pragma unroll
  for (int n = 0; n < D_STATE; ++n) h[n] = 0.f;

  for (int lc = 0; lc < SEQLEN; lc += 64) {
    __syncthreads();                          // previous chunk fully consumed
    // ---- async stage: 64 rows x 128B of (B,C) -> 512 x b128 segments
    {
      const float* srcbase = x_dbl + (size_t)(b * SEQLEN + lc) * 96 + 64;
#pragma unroll
      for (int rep = 0; rep < 2; ++rep) {
        int s = rep * 256 + threadIdx.x;      // segment id 0..511
        int t = s >> 3;                       // timestep in chunk
        int o = (s & 7) << 2;                 // float offset within 32-float row
        const float* g = srcbase + (size_t)t * 96 + o;
        unsigned lds_off = (unsigned)(uintptr_t)(&lds_bc[(t << 5) + o]);
        asm volatile("global_load_async_to_lds_b128 %0, %1, off"
                     :: "v"(lds_off), "v"(g) : "memory");
      }
    }
    WAIT_ASYNC();                             // s_wait_asynccnt 0
    __syncthreads();

    for (int t = 0; t < 64; ++t) {
      const int row   = b * SEQLEN + lc + t;
      const float dtt = dt[(size_t)row * D_INNER + d];
      const float ut  = u_act[(size_t)row * D_INNER + d];
      const float du  = dtt * ut;
      const float* Bt = &lds_bc[t * 32];
      const float* Ct = Bt + D_STATE;
      float y = 0.f;
#pragma unroll
      for (int n = 0; n < D_STATE; ++n) {
        h[n] = __expf(dtt * a[n]) * h[n] + du * Bt[n];
        y += h[n] * Ct[n];
      }
      y += ut * Dd;
      const float z = xz[(size_t)row * (2 * D_INNER) + D_INNER + d];
      y *= z * (1.0f / (1.0f + __expf(-z)));  // * silu(z)
      yg[(size_t)row * D_INNER + d] = (__bf16)y;
    }
  }
}

// ---------------------------------------------------------------------------
extern "C" void kernel_launch(void* const* d_in, const int* in_sizes, int n_in,
                              void* d_out, int out_size, void* d_ws, size_t ws_size,
                              hipStream_t stream) {
  const float* x      = (const float*)d_in[0];
  const float* w_in   = (const float*)d_in[1];
  const float* w_conv = (const float*)d_in[2];
  const float* b_conv = (const float*)d_in[3];
  const float* w_x    = (const float*)d_in[4];
  const float* w_dt   = (const float*)d_in[5];
  const float* b_dt   = (const float*)d_in[6];
  const float* A_log  = (const float*)d_in[7];
  const float* D_par  = (const float*)d_in[8];
  const float* w_out  = (const float*)d_in[9];
  float* out = (float*)d_out;

  char* ws = (char*)d_ws;
  __bf16* xb     = (__bf16*)(ws + XB_OFF);
  __bf16* w_inT  = (__bf16*)(ws + WINT_OFF);
  __bf16* w_outT = (__bf16*)(ws + WOUTT_OFF);
  float*  xz     = (float*)(ws + XZ_OFF);
  float*  u_act  = (float*)(ws + UACT_OFF);
  float*  x_dbl  = (float*)(ws + XDBL_OFF);
  float*  dtb    = (float*)(ws + DTB_OFF);
  __bf16* yg     = (__bf16*)(ws + YG_OFF);

  // 1) precision staging: x -> bf16, w_in/w_out -> transposed bf16
  k_cvt_bf16<<<(MROWS * D_MODEL + 255) / 256, 256, 0, stream>>>(x, xb, MROWS * D_MODEL);
  k_cvtT_bf16<<<(D_MODEL * 2 * D_INNER + 255) / 256, 256, 0, stream>>>(w_in, w_inT, D_MODEL, 2 * D_INNER);
  k_cvtT_bf16<<<(D_INNER * D_MODEL + 255) / 256, 256, 0, stream>>>(w_out, w_outT, D_INNER, D_MODEL);

  // 2) xz = x @ w_in : M=4096 N=4096 K=1024 -> (128 x 64) waves / 1024 blocks
  k_gemm_bf16_wmma<<<(MROWS / 32) * ((2 * D_INNER) / 64) / 8, 256, 0, stream>>>(
      xb, w_inT, xz, MROWS, 2 * D_INNER, D_MODEL);

  // 3) depthwise causal conv + SiLU
  k_conv_silu<<<(MROWS * D_INNER) / 256, 256, 0, stream>>>(xz, w_conv, b_conv, u_act);

  // 4) x_dbl = u @ w_x (N=96)
  k_xdbl<<<(MROWS * 96 + 255) / 256, 256, 0, stream>>>(u_act, w_x, x_dbl);

  // 5) dt = softplus(x_dbl[:, :64] @ w_dt + b_dt)
  k_dt<<<(MROWS * D_INNER) / 256, 256, 0, stream>>>(x_dbl, w_dt, b_dt, dtb);

  // 6) selective scan fused with skip/gating, bf16 out
  k_scan<<<BATCH * (D_INNER / 256), 256, 0, stream>>>(u_act, dtb, x_dbl, xz, A_log, D_par, yg);

  // 7) out = y_gated @ w_out : M=4096 N=1024 K=2048 -> (128 x 16) waves / 256 blocks
  k_gemm_bf16_wmma<<<(MROWS / 32) * (D_MODEL / 64) / 8, 256, 0, stream>>>(
      yg, w_outT, out, MROWS, D_MODEL, D_INNER);
}
